// BFA_57509612094011
// MI455X (gfx1250) — matrix operand order repaired and checked by reading, not compiled, not run on hardware
//
#include <hip/hip_runtime.h>
#include <hip/hip_bf16.h>

// ---------------------------------------------------------------------------
// CDNA5 (gfx1250) Restormer-style transposed-attention block.
// 1x1 convs = bf16 WMMA GEMMs (v_wmma_f32_16x16x32_bf16, f32 accumulate) with
// MT output tiles per wave so the activation (B) operand is loaded once and
// reused across MT WMMAs. 24x24 gram per (b,head) is also WMMA. Depthwise 3x3,
// LN, softmax, attn@V stay as bandwidth-optimal VALU kernels.
// ---------------------------------------------------------------------------

typedef __attribute__((ext_vector_type(16))) __bf16 v16bf;
typedef __attribute__((ext_vector_type(8)))  float  v8f;

static __device__ __forceinline__ __bf16 f2bf(float f) {
    union { float f; unsigned u; } a; a.f = f;
    unsigned r = a.u + 0x7FFFu + ((a.u >> 16) & 1u);   // round-to-nearest-even
    union { unsigned short s; __bf16 b; } c; c.s = (unsigned short)(r >> 16);
    return c.b;
}

// ---------------- weight convert: f32 [MxK] -> bf16 padded [Mp x Kp] --------
__global__ void k_convw(const float* __restrict__ src, __bf16* __restrict__ dst,
                        int M, int K, int Mp, int Kp) {
    int idx = blockIdx.x * 256 + threadIdx.x;
    if (idx >= Mp * Kp) return;
    int m = idx / Kp, k = idx % Kp;
    float v = (m < M && k < K) ? src[m * K + k] : 0.f;
    dst[idx] = f2bf(v);
}

// ---------------- channel LayerNorm (96 ch per pixel), f32 -> bf16 ----------
__global__ void k_ln(const float* __restrict__ x, const float* __restrict__ g,
                     const float* __restrict__ bta, __bf16* __restrict__ out) {
    int p  = blockIdx.x * 256 + threadIdx.x;   // 0..131071
    int b  = p >> 14, hw = p & 16383;
    const float* base = x + (long)b * 96 * 16384 + hw;
    float s = 0.f, ss = 0.f;
    for (int c = 0; c < 96; ++c) { float v = base[(long)c * 16384]; s += v; ss += v * v; }
    float mu  = s * (1.f / 96.f);
    float var = ss * (1.f / 96.f) - mu * mu;
    float inv = rsqrtf(var + 1e-5f);
    __bf16* ob = out + (long)b * 96 * 16384 + hw;
    for (int c = 0; c < 96; ++c) {
        float v = base[(long)c * 16384];
        ob[(long)c * 16384] = f2bf((v - mu) * inv * g[c] + bta[c]);
    }
}

// ---------------- WMMA GEMM, MT stacked M-tiles per wave --------------------
// out[b][m][n] (+res) = W[Mp x Kp] @ X[b][Kp][16384]. One wave per block;
// per K-step: one B fragment shared by MT WMMAs (B loaded once per MT tiles).
template <int MT>
__global__ void k_gemm(const __bf16* __restrict__ W, const __bf16* __restrict__ X,
                       float* __restrict__ out, const float* __restrict__ res,
                       int Kp, int Mreal, long strideXb, long strideOb) {
    const int lane = threadIdx.x;          // 32 threads (1 wave)
    const int half = lane >> 4, l15 = lane & 15;
    const int n0 = blockIdx.x * 16, m0 = blockIdx.y * (MT * 16), b = blockIdx.z;
    const __bf16* Xb = X + (long)b * strideXb;
    v8f acc[MT];
    #pragma unroll
    for (int t = 0; t < MT; ++t) acc[t] = v8f{};

    for (int ks = 0; ks < Kp; ks += 32) {
        // B tile 32x16: lane<16 col n0+lane holds K {ks..+7, ks+16..+23}; lane>=16 K {ks+8..15, ks+24..31}
        v16bf bm;
        const __bf16* xc = Xb + (long)(ks + half * 8) * 16384 + (n0 + l15);
        #pragma unroll
        for (int j = 0; j < 16; ++j) bm[j] = xc[(long)(j + (j & 8)) * 16384];
        #pragma unroll
        for (int t = 0; t < MT; ++t) {
            // A tile 16x32 for m-tile t: lane<16 row holds K {ks..+7, ks+16..+23};
            //                            lane>=16 same row, K {ks+8..15, ks+24..31}.
            const __bf16* wrow = W + (long)(m0 + t * 16 + l15) * Kp + ks;
            union { uint4 u; __bf16 h[8]; } ua, ub;
            ua.u = *(const uint4*)(wrow + half * 8);
            ub.u = *(const uint4*)(wrow + 16 + half * 8);
            v16bf a;
            #pragma unroll
            for (int j = 0; j < 8; ++j) { a[j] = ua.h[j]; a[8 + j] = ub.h[j]; }
            acc[t] = __builtin_amdgcn_wmma_f32_16x16x32_bf16(false, a, false, bm,
                                                             (short)0, acc[t], false, false);
        }
    }
    float* ob = out + (long)b * strideOb;
    const float* rb = res ? res + (long)b * strideOb : nullptr;
    #pragma unroll
    for (int t = 0; t < MT; ++t) {
        #pragma unroll
        for (int i = 0; i < 8; ++i) {
            int m = m0 + t * 16 + i + half * 8;
            if (m < Mreal) {
                long idx = (long)m * 16384 + (n0 + l15);
                float v = acc[t][i];
                if (rb) v += rb[idx];
                ob[idx] = v;
            }
        }
    }
}

// ---------------- depthwise 3x3 SAME, f32 -> f32 ----------------------------
__global__ void k_dw3(const float* __restrict__ in, const float* __restrict__ w9,
                      float* __restrict__ out, int C) {
    int hw = blockIdx.x * 256 + threadIdx.x;
    int c = blockIdx.y, b = blockIdx.z;
    int h = hw >> 7, x = hw & 127;
    const float* base = in + ((long)b * C + c) * 16384;
    const float* wt = w9 + c * 9;
    float s = 0.f;
    #pragma unroll
    for (int dy = -1; dy <= 1; ++dy) {
        int hh = h + dy; if ((unsigned)hh >= 128u) continue;
        #pragma unroll
        for (int dx = -1; dx <= 1; ++dx) {
            int ww = x + dx; if ((unsigned)ww >= 128u) continue;
            s += wt[(dy + 1) * 3 + dx + 1] * base[hh * 128 + ww];
        }
    }
    out[((long)b * C + c) * 16384 + hw] = s;
}

// ---------------- per-row 1/max(||.||,1e-12) over n=16384 -------------------
__global__ void k_rnorm(const float* __restrict__ qk, float* __restrict__ rn) {
    int row = blockIdx.x;                  // b*192 + ch
    const float* p = qk + (long)row * 16384;
    float s = 0.f;
    for (int i = threadIdx.x; i < 16384; i += 256) { float v = p[i]; s += v * v; }
    __shared__ float sm[256];
    sm[threadIdx.x] = s; __syncthreads();
    for (int o = 128; o; o >>= 1) { if (threadIdx.x < o) sm[threadIdx.x] += sm[threadIdx.x + o]; __syncthreads(); }
    if (threadIdx.x == 0) rn[row] = 1.f / fmaxf(sqrtf(sm[0]), 1e-12f);
}

// ---------------- WMMA gram: G[bh][32][32] = q(24xN) @ k(24xN)^T (padded) ---
__global__ void k_gram(const float* __restrict__ qk, float* __restrict__ G) {
    int lane = threadIdx.x;
    int half = lane >> 4, l15 = lane & 15;
    int bh = blockIdx.z, b = bh >> 2, h = bh & 3;
    const float* q = qk + ((long)b * 192 + h * 24) * 16384;
    const float* k = qk + ((long)b * 192 + 96 + h * 24) * 16384;
    int m0 = blockIdx.y * 16, n0 = blockIdx.x * 16;
    int mrow = m0 + l15, ncol = n0 + l15;
    bool mok = mrow < 24, nok = ncol < 24;
    v8f acc = {};
    const __bf16 z = f2bf(0.f);
    for (int ks = 0; ks < 16384; ks += 32) {
        v16bf a, bm;
        if (mok) {
            const float* qa = q + (long)mrow * 16384 + ks + half * 8;
            #pragma unroll
            for (int j = 0; j < 8; ++j) { a[j] = f2bf(qa[j]); a[8 + j] = f2bf(qa[16 + j]); }
        } else {
            #pragma unroll
            for (int j = 0; j < 16; ++j) a[j] = z;
        }
        if (nok) {
            const float* kc = k + (long)ncol * 16384 + ks + half * 8;
            #pragma unroll
            for (int j = 0; j < 16; ++j) bm[j] = f2bf(kc[j + (j & 8)]);
        } else {
            #pragma unroll
            for (int j = 0; j < 16; ++j) bm[j] = z;
        }
        acc = __builtin_amdgcn_wmma_f32_16x16x32_bf16(false, a, false, bm,
                                                      (short)0, acc, false, false);
    }
    float* g = G + (long)bh * 1024;
    #pragma unroll
    for (int i = 0; i < 8; ++i) g[(m0 + i + half * 8) * 32 + n0 + l15] = acc[i];
}

// ---------------- scale by rnorms & temperature, row softmax ----------------
__global__ void k_softmax(const float* __restrict__ G, const float* __restrict__ rn,
                          const float* __restrict__ temp, float* __restrict__ attn) {
    int bh = blockIdx.x, b = bh >> 2, h = bh & 3;
    int r = threadIdx.x; if (r >= 24) return;
    const float* g = G + (long)bh * 1024;
    float rq = rn[b * 192 + h * 24 + r], t = temp[h];
    float l[24], mx = -1e30f;
    for (int d = 0; d < 24; ++d) {
        float v = g[r * 32 + d] * rq * rn[b * 192 + 96 + h * 24 + d] * t;
        l[d] = v; mx = fmaxf(mx, v);
    }
    float s = 0.f;
    for (int d = 0; d < 24; ++d) { l[d] = __expf(l[d] - mx); s += l[d]; }
    float inv = 1.f / s;
    float* a = attn + (long)bh * 576 + r * 24;
    for (int d = 0; d < 24; ++d) a[d] = l[d] * inv;
}

// ---------------- out[b][96][n] = attn @ v, written bf16 --------------------
__global__ void k_attnv(const float* __restrict__ attn, const float* __restrict__ v,
                        __bf16* __restrict__ out) {
    int b = blockIdx.y;
    int hw = blockIdx.x * 256 + threadIdx.x;
    __shared__ float A[4 * 576];
    for (int i = threadIdx.x; i < 2304; i += 256) A[i] = attn[(long)b * 2304 + i];
    __syncthreads();
    const float* vb = v + (long)b * 96 * 16384 + hw;
    __bf16* ob = out + (long)b * 96 * 16384 + hw;
    for (int h = 0; h < 4; ++h) {
        float vr[24];
        #pragma unroll
        for (int d = 0; d < 24; ++d) vr[d] = vb[(long)(h * 24 + d) * 16384];
        const float* Ah = A + h * 576;
        #pragma unroll
        for (int c = 0; c < 24; ++c) {
            float s = 0.f;
            #pragma unroll
            for (int d = 0; d < 24; ++d) s += Ah[c * 24 + d] * vr[d];
            ob[(long)(h * 24 + c) * 16384] = f2bf(s);
        }
    }
}

// ---------------- fused dw3 + GELU gate: t[510] -> g bf16 [256 pad] ---------
__global__ void k_gelugate(const float* __restrict__ t, const float* __restrict__ w9,
                           __bf16* __restrict__ out) {
    int hw = blockIdx.x * 256 + threadIdx.x;
    int oc = blockIdx.y, b = blockIdx.z;
    __bf16* ob = out + ((long)b * 256 + oc) * 16384 + hw;
    if (oc >= 255) { *ob = f2bf(0.f); return; }   // K-pad row for 255->256
    int h = hw >> 7, x = hw & 127;
    const float* b1 = t + ((long)b * 510 + oc) * 16384;
    const float* b2 = t + ((long)b * 510 + oc + 255) * 16384;
    const float* w1 = w9 + oc * 9;
    const float* w2 = w9 + (oc + 255) * 9;
    float s1 = 0.f, s2 = 0.f;
    #pragma unroll
    for (int dy = -1; dy <= 1; ++dy) {
        int hh = h + dy; if ((unsigned)hh >= 128u) continue;
        #pragma unroll
        for (int dx = -1; dx <= 1; ++dx) {
            int ww = x + dx; if ((unsigned)ww >= 128u) continue;
            int wi = (dy + 1) * 3 + dx + 1, mi = hh * 128 + ww;
            s1 += w1[wi] * b1[mi];
            s2 += w2[wi] * b2[mi];
        }
    }
    float gelu = 0.5f * s1 * (1.f + erff(s1 * 0.70710678118654752f));
    *ob = f2bf(gelu * s2);
}

extern "C" void kernel_launch(void* const* d_in, const int* in_sizes, int n_in,
                              void* d_out, int out_size, void* d_ws, size_t ws_size,
                              hipStream_t stream) {
    (void)in_sizes; (void)n_in; (void)out_size; (void)ws_size;
    const float* x      = (const float*)d_in[0];
    const float* ln1_w  = (const float*)d_in[1];
    const float* ln1_b  = (const float*)d_in[2];
    const float* qk_w   = (const float*)d_in[3];
    const float* qk_dw  = (const float*)d_in[4];
    const float* v_w    = (const float*)d_in[5];
    const float* v_dw   = (const float*)d_in[6];
    const float* proj_w = (const float*)d_in[7];
    const float* temp   = (const float*)d_in[8];
    const float* ln2_w  = (const float*)d_in[9];
    const float* ln2_b  = (const float*)d_in[10];
    const float* pin_w  = (const float*)d_in[11];
    const float* ffn_dw = (const float*)d_in[12];
    const float* pout_w = (const float*)d_in[13];
    float* out = (float*)d_out;

    // ---- workspace layout (regions reused in stream order; ~546 MB total)
    char* ws = (char*)d_ws;
    size_t off = 0;
    __bf16* yb   = (__bf16*)(ws + off); off += 25165824;          // y / attnout / y2 (bf16)
    float*  r12  = (float*)(ws + off);                            // qkbuf | vbuf+vd | t
    float*  qkd  = (float*)(ws + off + 201326592);                // q,k after dw3
    float*  vbuf = r12;
    float*  vd   = (float*)(ws + off + 100663296);
    off += 402653184;
    float*  xa   = (float*)(ws + off); off += 50331648;           // attention-block output
    __bf16* gbuf = (__bf16*)(ws + off); off += 67108864;          // gated FFN act (256 pad)
    __bf16* wqk  = (__bf16*)(ws + off); off += 36864;
    __bf16* wv   = (__bf16*)(ws + off); off += 18432;
    __bf16* wpj  = (__bf16*)(ws + off); off += 18432;
    __bf16* wpi  = (__bf16*)(ws + off); off += 98304;             // 512x96
    __bf16* wpo  = (__bf16*)(ws + off); off += 49152;             // 96x256
    float*  rn   = (float*)(ws + off); off += 6144;
    float*  Gm   = (float*)(ws + off); off += 131072;             // 32x[32x32]
    float*  attn = (float*)(ws + off); off += 73728;

    const long SC96 = 96L * 16384, SC192 = 192L * 16384;
    const long SC510 = 510L * 16384, SC256 = 256L * 16384;

    // 1) weights -> padded bf16
    k_convw<<<(192 * 96 + 255) / 256, 256, 0, stream>>>(qk_w, wqk, 192, 96, 192, 96);
    k_convw<<<(96 * 96 + 255) / 256, 256, 0, stream>>>(v_w, wv, 96, 96, 96, 96);
    k_convw<<<(96 * 96 + 255) / 256, 256, 0, stream>>>(proj_w, wpj, 96, 96, 96, 96);
    k_convw<<<(512 * 96 + 255) / 256, 256, 0, stream>>>(pin_w, wpi, 510, 96, 512, 96);
    k_convw<<<(96 * 256 + 255) / 256, 256, 0, stream>>>(pout_w, wpo, 96, 255, 96, 256);

    // 2) LN1 -> bf16
    k_ln<<<512, 256, 0, stream>>>(x, ln1_w, ln1_b, yb);
    // 3) qk = qk_w @ y (WMMA, 2 wave-groups of 6 m-tiles), then dw3
    k_gemm<6><<<dim3(1024, 2, 8), 32, 0, stream>>>(wqk, yb, r12, nullptr, 96, 192, SC96, SC192);
    k_dw3<<<dim3(64, 192, 8), 256, 0, stream>>>(r12, qk_dw, qkd, 192);
    // 4) v = v_w @ y (WMMA), then dw3
    k_gemm<6><<<dim3(1024, 1, 8), 32, 0, stream>>>(wv, yb, vbuf, nullptr, 96, 96, SC96, SC96);
    k_dw3<<<dim3(64, 96, 8), 256, 0, stream>>>(vbuf, v_dw, vd, 96);
    // 5) row norms, gram (WMMA), softmax, attn@v
    k_rnorm<<<1536, 256, 0, stream>>>(qkd, rn);
    k_gram<<<dim3(2, 2, 32), 32, 0, stream>>>(qkd, Gm);
    k_softmax<<<32, 32, 0, stream>>>(Gm, rn, temp, attn);
    k_attnv<<<dim3(64, 8), 256, 0, stream>>>(attn, vd, yb);
    // 6) xa = x + proj_w @ attnout (WMMA + residual)
    k_gemm<6><<<dim3(1024, 1, 8), 32, 0, stream>>>(wpj, yb, xa, x, 96, 96, SC96, SC96);
    // 7) LN2 -> bf16
    k_ln<<<512, 256, 0, stream>>>(xa, ln2_w, ln2_b, yb);
    // 8) t = pin_w @ y2 (WMMA, 4 wave-groups of 8 m-tiles, 510 real rows)
    k_gemm<8><<<dim3(1024, 4, 8), 32, 0, stream>>>(wpi, yb, r12, nullptr, 96, 510, SC96, SC510);
    // 9) fused dw3 + gelu gate -> bf16 (with zero pad row 255)
    k_gelugate<<<dim3(64, 256, 8), 256, 0, stream>>>(r12, ffn_dw, gbuf);
    // 10) out = xa + pout_w @ g (WMMA + residual)
    k_gemm<6><<<dim3(1024, 1, 8), 32, 0, stream>>>(wpo, gbuf, out, xa, 256, 96, SC256, SC96);
}
